// GET_59931973649049
// MI455X (gfx1250) — matrix operand order repaired and checked by reading, not compiled, not run on hardware
//
#include <hip/hip_runtime.h>
#include <hip/hip_bf16.h>

typedef __attribute__((ext_vector_type(16))) _Float16 v16h;
typedef __attribute__((ext_vector_type(8)))  float    v8f;

#define NBATCH 8

// ---------------- parameter offsets (floats), pytree sorted-dict order ----
// attn-relative
#define ATT0_W 0
#define ATT0_B 832
#define ATT1_W 848
#define ATT1_B 1104
#define BA_W  1120
#define BA_B  1136
#define C0_W  1137
#define C0_B  9329
#define C1_W  9457
#define C1_B  9969
#define E_W   9973
#define E_B   10997
#define N0_W  11061
#define N0_B  19253
#define N1_W  19381
#define N1_B  35765
#define NO_W  35893
#define NO_B  52277
#define QK_W  52405
#define UM0_W 53429
#define UM0_B 53749
#define UM1_W 53765
#define UM1_B 54021
#define UC0_W 54037
#define UC0_B 54357
#define UC1_W 54373
#define UC1_B 54629
#define UCL_W 54645
#define UCL_B 54661
#define UA_W  54662
#define UA_B  54678
#define V_W   54679
#define V_B   55191
#define ATTN_SZ 55207
// ffn-relative
#define H0_W 0
#define H0_B 69632
#define H1_W 69888
#define H1_B 135424
#define H2_W 135680
#define H2_B 168448
#define Z0_W 168576
#define Z0_B 238208
#define Z1_W 238464
#define Z1_B 304000
#define Z2_W 304256
#define Z2_B 304512
#define FFN_SZ 304513
// ln-relative
#define F0_W 0
#define F0_B 2048
#define F1_W 2176
#define F1_B 18560
#define LNB  18688
#define LNG  18816
#define SIG  18944
#define LN_SZ 18945
#define LAYER_SZ (ATTN_SZ + FFN_SZ + 2 * LN_SZ)

// ---------------- device helpers -----------------------------------------
__device__ __forceinline__ float silu(float x) { return x / (1.f + __expf(-x)); }

__device__ __forceinline__ void rbf16(float d, float* out) {
  const float p = 5.f;
  float ds = d * (1.f / 7.0f);
  float ds2 = ds * ds, ds4 = ds2 * ds2, ds5 = ds4 * ds, ds6 = ds5 * ds, ds7 = ds6 * ds;
  float env = 1.f - 0.5f * (p + 1.f) * (p + 2.f) * ds5 + p * (p + 2.f) * ds6
            - 0.5f * p * (p + 1.f) * ds7;
  env = (ds < 1.f) ? env : 0.f;
  const float coeff = -112.5f;            // -0.5/(1/15)^2
  const float step = 1.f / 15.f;
#pragma unroll
  for (int i = 0; i < 16; ++i) {
    float t = ds - (float)i * step;
    out[i] = env * __expf(coeff * t * t);
  }
}

// WMMA fragment loaders from row-major LDS tiles (contiguous k = 16*(lane/16) + i for both A and B).
__device__ __forceinline__ v16h load_a_frag(const _Float16* tile, int ld, int lane) {
  v16h a;
  int m = lane & 15;
  int kb = (lane >= 16) ? 16 : 0;
#pragma unroll
  for (int i = 0; i < 16; ++i) {
    int k = i + kb;
    a[i] = tile[m * ld + k];
  }
  return a;
}
__device__ __forceinline__ v16h load_b_frag(const _Float16* tile, int ld, int lane) {
  v16h b;
  int n = lane & 15;
  int kb = (lane >= 16) ? 16 : 0;
#pragma unroll
  for (int i = 0; i < 16; ++i) b[i] = tile[(kb + i) * ld + n];
  return b;
}

__device__ __forceinline__ unsigned fenc(float f) {
  unsigned u = __float_as_uint(f);
  return (u & 0x80000000u) ? ~u : (u | 0x80000000u);
}
__device__ __forceinline__ float fdec(unsigned u) {
  unsigned b = (u & 0x80000000u) ? (u & 0x7fffffffu) : ~u;
  return __uint_as_float(b);
}

// ---------------- generic WMMA GEMM: C = [resid +] act(A[M,K] @ W[K,N] + b)
__global__ void __launch_bounds__(32)
gemm_kernel(const float* __restrict__ A, const float* __restrict__ W,
            const float* __restrict__ bias, const float* __restrict__ resid,
            float* __restrict__ C, int M, int K, int N, int act) {
  __shared__ _Float16 At[16 * 32];
  __shared__ _Float16 Wt[32 * 16];
  int tileM = blockIdx.x * 16, tileN = blockIdx.y * 16;
  int lane = threadIdx.x;
  v8f acc = {0.f, 0.f, 0.f, 0.f, 0.f, 0.f, 0.f, 0.f};
  for (int k0 = 0; k0 < K; k0 += 32) {
    for (int i = lane; i < 512; i += 32) {
      int r = i >> 5, c = i & 31;
      int gm = tileM + r, gk = k0 + c;
      float v = (gm < M && gk < K) ? A[(size_t)gm * K + gk] : 0.f;
      At[r * 32 + c] = (_Float16)v;
    }
    for (int i = lane; i < 512; i += 32) {
      int r = i >> 4, c = i & 15;
      int gk = k0 + r, gn = tileN + c;
      float v = (gk < K && gn < N) ? W[(size_t)gk * N + gn] : 0.f;
      Wt[r * 16 + c] = (_Float16)v;
    }
    __syncthreads();
    v16h a = load_a_frag(At, 32, lane);
    v16h b = load_b_frag(Wt, 16, lane);
    acc = __builtin_amdgcn_wmma_f32_16x16x32_f16(false, a, false, b, (short)0, acc,
                                                 false, false);
    __syncthreads();
  }
  int n = lane & 15, gn = tileN + n;
  int mb = (lane >= 16) ? 8 : 0;
  float bv = (bias != nullptr && gn < N) ? bias[gn] : 0.f;
#pragma unroll
  for (int r = 0; r < 8; ++r) {
    int gm = tileM + mb + r;
    if (gm < M && gn < N) {
      float v = acc[r] + bv;
      if (act) v = silu(v);
      if (resid) v += resid[(size_t)gm * N + gn];
      C[(size_t)gm * N + gn] = v;
    }
  }
}

// ---------------- fused per-block-edge attention kernel -------------------
// 1 block edge per workgroup: 16 unit edges x 4 heads = 64 rows.
__global__ void __launch_bounds__(128)
edge_kernel(const float* __restrict__ Hqk, const float* __restrict__ Hv,
            const float* __restrict__ ea, const float* __restrict__ Z,
            const int* __restrict__ esrc, const int* __restrict__ edst,
            const float* __restrict__ attn, float* __restrict__ Rmean,
            float* __restrict__ HaggIn, float* __restrict__ ZHin,
            float* __restrict__ ZaggG, int E) {
  int e = blockIdx.x;
  int t = threadIdx.x;
  int lane = t & 31, wave = t >> 5;
  int src = esrc[e], dst = edst[e];

  __shared__ _Float16 A1[64 * 64];       // att input rows, K pad 52->64
  __shared__ _Float16 A2[64 * 32];       // msg input rows, K pad 20->32
  __shared__ _Float16 W1a[64 * 16];
  __shared__ _Float16 W2a[32 * 16];
  __shared__ _Float16 W1m[32 * 16];
  __shared__ _Float16 W2m[32 * 16];
  __shared__ _Float16 W1c[32 * 16];
  __shared__ _Float16 W2c[32 * 16];
  __shared__ _Float16 Htmp[4][16 * 32];  // per-wave restage (zero-padded K)
  __shared__ float rbfv[16][16];
  __shared__ float dZl[16][3];
  __shared__ float Rrep[64][16];
  __shared__ float Hm[64][16];
  __shared__ float Hcs[64][16];
  __shared__ float Rl[64], wll[64], alphal[64];

  // stage weights (f32 -> f16)
  for (int i = t; i < 64 * 16; i += 128) {
    int r = i >> 4, c = i & 15;
    W1a[i] = (_Float16)(r < 52 ? attn[ATT0_W + r * 16 + c] : 0.f);
  }
  for (int i = t; i < 32 * 16; i += 128) {
    int r = i >> 4, c = i & 15;
    W2a[i] = (_Float16)(r < 16 ? attn[ATT1_W + r * 16 + c] : 0.f);
    W1m[i] = (_Float16)(r < 20 ? attn[UM0_W + r * 16 + c] : 0.f);
    W2m[i] = (_Float16)(r < 16 ? attn[UM1_W + r * 16 + c] : 0.f);
    W1c[i] = (_Float16)(r < 20 ? attn[UC0_W + r * 16 + c] : 0.f);
    W2c[i] = (_Float16)(r < 16 ? attn[UC1_W + r * 16 + c] : 0.f);
  }
  // per unit-edge geometry + rbf
  if (t < 16) {
    int j = t >> 2, jj = t & 3;
    int rn = src * 4 + j, cn = dst * 4 + jj;
    float d0 = Z[rn * 3 + 0] - Z[cn * 3 + 0];
    float d1 = Z[rn * 3 + 1] - Z[cn * 3 + 1];
    float d2 = Z[rn * 3 + 2] - Z[cn * 3 + 2];
    dZl[t][0] = d0; dZl[t][1] = d1; dZl[t][2] = d2;
    float r16[16];
    rbf16(sqrtf(d0 * d0 + d1 * d1 + d2 * d2), r16);
#pragma unroll
    for (int i = 0; i < 16; ++i) rbfv[t][i] = r16[i];
  }
  __syncthreads();

  // build A1 (52-wide att input) and A2 (20-wide msg input)
  for (int i = t; i < 64 * 64; i += 128) {
    int row = i >> 6, col = i & 63;
    int ue = row >> 2, h = row & 3, j = ue >> 2, jj = ue & 3;
    float v = 0.f;
    if (col < 16)       v = Hqk[(size_t)(src * 4 + j) * 128 + h * 32 + 2 * col];
    else if (col < 32)  v = Hqk[(size_t)(dst * 4 + jj) * 128 + h * 32 + 2 * (col - 16) + 1];
    else if (col < 36)  v = rbfv[ue][h * 4 + (col - 32)];
    else if (col < 52)  v = ea[(size_t)e * 64 + h * 16 + (col - 36)];
    A1[i] = (_Float16)v;
  }
  for (int i = t; i < 64 * 32; i += 128) {
    int row = i >> 5, col = i & 31;
    int ue = row >> 2, h = row & 3, jj = ue & 3;
    float v = 0.f;
    if (col < 16)      v = Hv[(size_t)(dst * 4 + jj) * 64 + h * 16 + col];
    else if (col < 20) v = rbfv[ue][h * 4 + (col - 16)];
    A2[i] = (_Float16)v;
  }
  __syncthreads();

  int n = lane & 15, mb = (lane >= 16) ? 8 : 0;

  // ---- chain ATT: silu(silu(A1@W1a+b)@W2a+b) -> Rrep
  {
    const _Float16* Ab = A1 + wave * 16 * 64;
    v8f acc = {0.f, 0.f, 0.f, 0.f, 0.f, 0.f, 0.f, 0.f};
    acc = __builtin_amdgcn_wmma_f32_16x16x32_f16(false, load_a_frag(Ab, 64, lane), false,
                                                 load_b_frag(W1a, 16, lane), (short)0,
                                                 acc, false, false);
    acc = __builtin_amdgcn_wmma_f32_16x16x32_f16(false, load_a_frag(Ab + 32, 64, lane),
                                                 false, load_b_frag(W1a + 32 * 16, 16, lane),
                                                 (short)0, acc, false, false);
    float bv = attn[ATT0_B + n];
    _Float16* tp = &Htmp[wave][0];
#pragma unroll
    for (int r = 0; r < 8; ++r) {
      tp[(mb + r) * 32 + n] = (_Float16)silu(acc[r] + bv);
      tp[(mb + r) * 32 + 16 + n] = (_Float16)0.f;
    }
  }
  __syncthreads();
  {
    v8f acc = {0.f, 0.f, 0.f, 0.f, 0.f, 0.f, 0.f, 0.f};
    acc = __builtin_amdgcn_wmma_f32_16x16x32_f16(false, load_a_frag(&Htmp[wave][0], 32, lane),
                                                 false, load_b_frag(W2a, 16, lane), (short)0,
                                                 acc, false, false);
    float bv = attn[ATT1_B + n];
#pragma unroll
    for (int r = 0; r < 8; ++r) Rrep[wave * 16 + mb + r][n] = silu(acc[r] + bv);
  }
  __syncthreads();

  // ---- chain MSG (umsg): A2 -> Hm
  {
    const _Float16* Ab = A2 + wave * 16 * 32;
    v8f acc = {0.f, 0.f, 0.f, 0.f, 0.f, 0.f, 0.f, 0.f};
    acc = __builtin_amdgcn_wmma_f32_16x16x32_f16(false, load_a_frag(Ab, 32, lane), false,
                                                 load_b_frag(W1m, 16, lane), (short)0,
                                                 acc, false, false);
    float bv = attn[UM0_B + n];
    _Float16* tp = &Htmp[wave][0];
#pragma unroll
    for (int r = 0; r < 8; ++r) {
      tp[(mb + r) * 32 + n] = (_Float16)silu(acc[r] + bv);
      tp[(mb + r) * 32 + 16 + n] = (_Float16)0.f;
    }
  }
  __syncthreads();
  {
    v8f acc = {0.f, 0.f, 0.f, 0.f, 0.f, 0.f, 0.f, 0.f};
    acc = __builtin_amdgcn_wmma_f32_16x16x32_f16(false, load_a_frag(&Htmp[wave][0], 32, lane),
                                                 false, load_b_frag(W2m, 16, lane), (short)0,
                                                 acc, false, false);
    float bv = attn[UM1_B + n];
#pragma unroll
    for (int r = 0; r < 8; ++r) Hm[wave * 16 + mb + r][n] = silu(acc[r] + bv);
  }
  __syncthreads();

  // ---- chain MSGC (umsgc): A2 -> Hcs
  {
    const _Float16* Ab = A2 + wave * 16 * 32;
    v8f acc = {0.f, 0.f, 0.f, 0.f, 0.f, 0.f, 0.f, 0.f};
    acc = __builtin_amdgcn_wmma_f32_16x16x32_f16(false, load_a_frag(Ab, 32, lane), false,
                                                 load_b_frag(W1c, 16, lane), (short)0,
                                                 acc, false, false);
    float bv = attn[UC0_B + n];
    _Float16* tp = &Htmp[wave][0];
#pragma unroll
    for (int r = 0; r < 8; ++r) {
      tp[(mb + r) * 32 + n] = (_Float16)silu(acc[r] + bv);
      tp[(mb + r) * 32 + 16 + n] = (_Float16)0.f;
    }
  }
  __syncthreads();
  {
    v8f acc = {0.f, 0.f, 0.f, 0.f, 0.f, 0.f, 0.f, 0.f};
    acc = __builtin_amdgcn_wmma_f32_16x16x32_f16(false, load_a_frag(&Htmp[wave][0], 32, lane),
                                                 false, load_b_frag(W2c, 16, lane), (short)0,
                                                 acc, false, false);
    float bv = attn[UC1_B + n];
#pragma unroll
    for (int r = 0; r < 8; ++r) Hcs[wave * 16 + mb + r][n] = silu(acc[r] + bv);
  }
  __syncthreads();

  // per-row scalars: R = Rrep.unit_att, wlin = Hcs.umsgc_lin
  if (t < 64) {
    float r = attn[UA_B], w2 = attn[UCL_B];
#pragma unroll
    for (int i = 0; i < 16; ++i) {
      r  += Rrep[t][i] * attn[UA_W + i];
      w2 += Hcs[t][i] * attn[UCL_W + i];
    }
    Rl[t] = r;
    wll[t] = w2;
  }
  __syncthreads();
  // alpha: softmax over jj within (j,h) group (rows (j*4+jj)*4+h)
  if (t < 64) {
    int h = t & 3, ue = t >> 2, j = ue >> 2;
    float m = -3.0e38f;
#pragma unroll
    for (int jj = 0; jj < 4; ++jj) m = fmaxf(m, Rl[(j * 4 + jj) * 4 + h]);
    float s = 0.f;
#pragma unroll
    for (int jj = 0; jj < 4; ++jj) s += __expf(Rl[(j * 4 + jj) * 4 + h] - m);
    alphal[t] = __expf(Rl[t] - m) / s;
  }
  __syncthreads();

  // outputs
  if (t < 64) {  // Rmean[e][h][c]
    int h = t >> 4, c = t & 15;
    float s = 0.f;
#pragma unroll
    for (int ue = 0; ue < 16; ++ue) s += Rrep[ue * 4 + h][c];
    Rmean[((size_t)e * 4 + h) * 16 + c] = s * (1.f / 16.f);
  }
  for (int idx = t; idx < 256; idx += 128) {  // HaggIn/ZHin [Gu][h*16+c]
    int j = idx >> 6, rem = idx & 63, h = rem >> 4, c = rem & 15;
    float sh = 0.f, sc = 0.f;
#pragma unroll
    for (int jj = 0; jj < 4; ++jj) {
      int row = (j * 4 + jj) * 4 + h;
      float al = alphal[row];
      sh += al * Hm[row][c];
      sc += al * Hcs[row][c];
    }
    size_t g = (size_t)e * 4 + j;
    HaggIn[g * 64 + h * 16 + c] = sh;
    ZHin[g * 64 + h * 16 + c] = sc;
  }
  if (t < 48) {  // ZaggG[g][h][d] = sum_jj alpha*wlin*dZ
    int j = t / 12, rem = t % 12, h = rem / 3, d = rem % 3;
    float s = 0.f;
#pragma unroll
    for (int jj = 0; jj < 4; ++jj) {
      int ue = j * 4 + jj, row = ue * 4 + h;
      s += alphal[row] * wll[row] * dZl[ue][d];
    }
    ZaggG[(((size_t)e * 4 + j) * 4 + h) * 3 + d] = s;
  }
}

// ---------------- beta (block softmax) kernels ----------------------------
__global__ void k_blogit(const float* __restrict__ Rmean, const float* __restrict__ attn,
                         float* __restrict__ blogit, int E) {
  int i = blockIdx.x * 256 + threadIdx.x;
  if (i >= E * 4) return;
  const float* r = Rmean + (size_t)i * 16;
  float s = attn[BA_B];
#pragma unroll
  for (int k = 0; k < 16; ++k) s += r[k] * attn[BA_W + k];
  blogit[i] = s;
}
__global__ void k_initmax(unsigned* p, int n) {
  int i = blockIdx.x * 256 + threadIdx.x;
  if (i < n) p[i] = fenc(-3.0e38f);
}
__global__ void k_bmax(const float* __restrict__ blogit, const int* __restrict__ esrc,
                       unsigned* __restrict__ bmax, int E) {
  int i = blockIdx.x * 256 + threadIdx.x;
  if (i >= E * 4) return;
  int e = i >> 2, h = i & 3;
  atomicMax(&bmax[esrc[e] * 4 + h], fenc(blogit[i]));
}
__global__ void k_bexp(float* __restrict__ blogit, const int* __restrict__ esrc,
                       const unsigned* __restrict__ bmax, float* __restrict__ bsum, int E) {
  int i = blockIdx.x * 256 + threadIdx.x;
  if (i >= E * 4) return;
  int e = i >> 2, h = i & 3;
  float v = __expf(blogit[i] - fdec(bmax[esrc[e] * 4 + h]));
  blogit[i] = v;
  atomicAdd(&bsum[esrc[e] * 4 + h], v);
}
__global__ void k_bnorm(float* __restrict__ blogit, const int* __restrict__ esrc,
                        const float* __restrict__ bsum, int E) {
  int i = blockIdx.x * 256 + threadIdx.x;
  if (i >= E * 4) return;
  int e = i >> 2, h = i & 3;
  blogit[i] /= bsum[esrc[e] * 4 + h];
}

// ---------------- scatter to nodes ---------------------------------------
__global__ void k_scatterH(const float* __restrict__ NodeOutG, const float* __restrict__ beta,
                           const int* __restrict__ esrc, float* __restrict__ HaggN, int Gu) {
  size_t i = (size_t)blockIdx.x * 256 + threadIdx.x;
  if (i >= (size_t)Gu * 128) return;
  int g = (int)(i >> 7), f = (int)(i & 127), h = f >> 5;
  int e = g >> 2, j = g & 3;
  int node = esrc[e] * 4 + j;
  atomicAdd(&HaggN[(size_t)node * 128 + f], beta[e * 4 + h] * NodeOutG[(size_t)g * 128 + f]);
}
__global__ void k_scatterZ(const float* __restrict__ ZaggG, const float* __restrict__ ZHG,
                           const float* __restrict__ beta, const int* __restrict__ esrc,
                           float* __restrict__ ZaggN, int Gu) {
  int i = blockIdx.x * 256 + threadIdx.x;
  if (i >= Gu * 12) return;
  int g = i / 12, rem = i % 12, h = rem / 3, d = rem % 3;
  int e = g >> 2, j = g & 3;
  int node = esrc[e] * 4 + j;
  atomicAdd(&ZaggN[((size_t)node * 4 + h) * 3 + d],
            beta[e * 4 + h] * ZHG[(size_t)g * 4 + h] * ZaggG[((size_t)g * 4 + h) * 3 + d]);
}
__global__ void k_zupd(const float* __restrict__ Z, const float* __restrict__ ZaggN,
                       float* __restrict__ Zn, int N) {
  int i = blockIdx.x * 256 + threadIdx.x;
  if (i >= N * 3) return;
  int nn = i / 3, d = i % 3;
  float s = Z[i];
#pragma unroll
  for (int h = 0; h < 4; ++h) s += ZaggN[((size_t)nn * 4 + h) * 3 + d];
  Zn[i] = s;
}

// ---------------- eq_layernorm kernels ------------------------------------
__global__ void k_zmean(const float* __restrict__ Z, float* __restrict__ Zbm, int N) {
  int b = blockIdx.x, t = threadIdx.x;
  int npb = N / NBATCH, base = b * npb;
  __shared__ float red[256];
  for (int d = 0; d < 3; ++d) {
    float s = 0.f;
    for (int i = t; i < npb; i += 256) s += Z[(size_t)(base + i) * 3 + d];
    red[t] = s;
    __syncthreads();
    for (int o = 128; o > 0; o >>= 1) {
      if (t < o) red[t] += red[t + o];
      __syncthreads();
    }
    if (t == 0) Zbm[b * 3 + d] = red[0] / (float)npb;
    __syncthreads();
  }
}
__global__ void k_std(const float* __restrict__ Z, const float* __restrict__ Zbm,
                      const float* __restrict__ lnp, float* __restrict__ rescale,
                      float* __restrict__ rrb, int N) {
  int b = blockIdx.x, t = threadIdx.x;
  int npb = N / NBATCH, base = b * npb;
  __shared__ float red[256];
  float s = 0.f;
  for (int i = t; i < npb * 3; i += 256) {
    int node = base + i / 3, d = i % 3;
    float zc = Z[(size_t)node * 3 + d] - Zbm[b * 3 + d];
    s += zc * zc;
  }
  red[t] = s;
  __syncthreads();
  for (int o = 128; o > 0; o >>= 1) {
    if (t < o) red[t] += red[t + o];
    __syncthreads();
  }
  if (t == 0) {
    float var = red[0] / (float)(npb * 3 - 1);
    float resc = lnp[SIG] / sqrtf(var);
    rescale[b] = resc;
    float r16[16];
    rbf16(resc, r16);
    for (int i = 0; i < 16; ++i) rrb[b * 16 + i] = r16[i];
  }
}
__global__ void __launch_bounds__(128)
k_lnapply(const float* __restrict__ H, const float* __restrict__ Z,
          const float* __restrict__ fuse_o, const float* __restrict__ Zbm,
          const float* __restrict__ rescale, const float* __restrict__ lnp,
          float* __restrict__ Hn, float* __restrict__ Zn, int N) {
  int nn = blockIdx.x, t = threadIdx.x;
  int b = nn / (N / NBATCH);
  __shared__ float red[128];
  __shared__ float mu_s, var_s;
  float hv = H[(size_t)nn * 128 + t] + fuse_o[b * 128 + t];
  red[t] = hv;
  __syncthreads();
  for (int o = 64; o > 0; o >>= 1) {
    if (t < o) red[t] += red[t + o];
    __syncthreads();
  }
  if (t == 0) mu_s = red[0] * (1.f / 128.f);
  __syncthreads();
  float d = hv - mu_s;
  red[t] = d * d;
  __syncthreads();
  for (int o = 64; o > 0; o >>= 1) {
    if (t < o) red[t] += red[t + o];
    __syncthreads();
  }
  if (t == 0) var_s = red[0] * (1.f / 128.f);
  __syncthreads();
  Hn[(size_t)nn * 128 + t] = d * rsqrtf(var_s + 1e-5f) * lnp[LNG + t] + lnp[LNB + t];
  if (t < 3) {
    float zc = Z[(size_t)nn * 3 + t] - Zbm[b * 3 + t];
    Zn[(size_t)nn * 3 + t] = Zbm[b * 3 + t] + zc * rescale[b];
  }
}

// ---------------- eq_ffn kernels ------------------------------------------
__global__ void __launch_bounds__(128)
k_blockmean(const float* __restrict__ H, const float* __restrict__ Z,
            float* __restrict__ Hblk, float* __restrict__ Zblk) {
  int blk = blockIdx.x, t = threadIdx.x;
  float s = 0.f;
#pragma unroll
  for (int i = 0; i < 4; ++i) s += H[((size_t)blk * 4 + i) * 128 + t];
  Hblk[(size_t)blk * 128 + t] = s * 0.25f;
  if (t < 3) {
    float sz = 0.f;
#pragma unroll
    for (int i = 0; i < 4; ++i) sz += Z[((size_t)blk * 4 + i) * 3 + t];
    Zblk[blk * 3 + t] = sz * 0.25f;
  }
}
__global__ void __launch_bounds__(64)
k_ffnin(const float* __restrict__ H, const float* __restrict__ Z,
        const float* __restrict__ Hblk, const float* __restrict__ Zblk,
        float* __restrict__ inX, int N) {
  int nn = blockIdx.x, t = threadIdx.x;
  int blk = nn >> 2;
  __shared__ float rad[16];
  if (t == 0) {
    float s2 = 0.f;
    for (int d = 0; d < 3; ++d) {
      float zo = Z[(size_t)nn * 3 + d] - Zblk[blk * 3 + d];
      s2 += zo * zo;
    }
    float r16[16];
    rbf16(sqrtf(s2), r16);
    for (int i = 0; i < 16; ++i) rad[i] = r16[i];
  }
  __syncthreads();
  for (int c = t; c < 272; c += 64) {
    float v;
    if (c < 128) v = H[(size_t)nn * 128 + c];
    else if (c < 256) v = Hblk[(size_t)blk * 128 + (c - 128)];
    else v = rad[c - 256];
    inX[(size_t)nn * 272 + c] = v;
  }
}
__global__ void k_zscal(const float* __restrict__ t256, const float* __restrict__ ffnp,
                        const float* __restrict__ Z, const float* __restrict__ Zblk,
                        float* __restrict__ Zn, int N) {
  int nn = blockIdx.x * 256 + threadIdx.x;
  if (nn >= N) return;
  float s = ffnp[Z2_B];
  const float* r = t256 + (size_t)nn * 256;
  for (int k = 0; k < 256; ++k) s += r[k] * ffnp[Z2_W + k];
  int blk = nn >> 2;
  for (int d = 0; d < 3; ++d) {
    float zc = Z[(size_t)nn * 3 + d] - Zblk[blk * 3 + d];
    Zn[(size_t)nn * 3 + d] = Zblk[blk * 3 + d] + s * zc;
  }
}

// ---------------- host orchestration --------------------------------------
extern "C" void kernel_launch(void* const* d_in, const int* in_sizes, int n_in,
                              void* d_out, int out_size, void* d_ws, size_t ws_size,
                              hipStream_t stream) {
  const float* Hin = (const float*)d_in[0];
  const float* Zin = (const float*)d_in[1];
  const float* EAin = (const float*)d_in[2];
  const float* P = (const float*)d_in[3];
  const int* edges = (const int*)d_in[6];

  int N = in_sizes[0] / 128;     // 8000
  int E = in_sizes[2] / 16;      // 24000
  int Gu = E * 4;
  int NBLK = N / 4;
  const int* esrc = edges;
  const int* edst = edges + E;

  float* w = (float*)d_ws;
  size_t o = 0;
  auto alloc = [&](size_t cnt) { float* p = w + o; o += cnt; return p; };
  float* Ha = alloc((size_t)N * 128);
  float* Hb = alloc((size_t)N * 128);
  float* Za = alloc((size_t)N * 3);
  float* Zb = alloc((size_t)N * 3);
  float* Hqk = alloc((size_t)N * 128);
  float* Hv = alloc((size_t)N * 64);
  float* ea = alloc((size_t)E * 64);
  float* Rmean = alloc((size_t)E * 64);
  float* HaggIn = alloc((size_t)Gu * 64);
  float* ZHin = alloc((size_t)Gu * 64);
  float* ZaggG = alloc((size_t)Gu * 12);
  float* T1 = alloc((size_t)Gu * 128);
  float* NodeOutG = alloc((size_t)Gu * 128);
  float* ZHG = alloc((size_t)Gu * 4);
  float* blogit = alloc((size_t)E * 4);
  unsigned* bmax = (unsigned*)alloc((size_t)NBLK * 4);
  float* bsum = alloc((size_t)NBLK * 4);
  float* HaggN = alloc((size_t)N * 128);
  float* ZaggN = alloc((size_t)N * 12);
  float* Zbm = alloc(NBATCH * 3);
  float* rescB = alloc(NBATCH);
  float* rrb = alloc(NBATCH * 16);
  float* fuse_h = alloc(NBATCH * 128);
  float* fuse_o = alloc(NBATCH * 128);
  float* Hblk = alloc((size_t)NBLK * 128);
  float* Zblk = alloc((size_t)NBLK * 3);
  // FFN-phase buffers alias dead attention-phase scratch
  float* in272 = HaggIn;
  float* t256a = ZHin;
  float* t256b = T1;
  (void)ws_size; (void)n_in; (void)out_size;

  hipMemcpyAsync(Ha, Hin, (size_t)N * 128 * sizeof(float), hipMemcpyDeviceToDevice, stream);
  hipMemcpyAsync(Za, Zin, (size_t)N * 3 * sizeof(float), hipMemcpyDeviceToDevice, stream);

  auto gemm = [&](const float* A, const float* W, const float* b, const float* resid,
                  float* C, int M, int K, int Nn, int act) {
    dim3 g((M + 15) / 16, (Nn + 15) / 16);
    gemm_kernel<<<g, 32, 0, stream>>>(A, W, b, resid, C, M, K, Nn, act);
  };

  float* curH = Ha; float* nxtH = Hb; float* curZ = Za; float* nxtZ = Zb;

  for (int l = 0; l < 2; ++l) {
    const float* attn = P + (size_t)l * LAYER_SZ;
    const float* ffnp = attn + ATTN_SZ;
    const float* ln1p = ffnp + FFN_SZ;
    const float* ln2p = ln1p + LN_SZ;

    // ---- attention ----
    gemm(curH, attn + QK_W, nullptr, nullptr, Hqk, 4 * N, 32, 32, 0);
    gemm(curH, attn + V_W, attn + V_B, nullptr, Hv, 4 * N, 32, 16, 0);
    gemm(EAin, attn + E_W, attn + E_B, nullptr, ea, E, 16, 64, 0);
    edge_kernel<<<E, 128, 0, stream>>>(Hqk, Hv, ea, curZ, esrc, edst, attn,
                                       Rmean, HaggIn, ZHin, ZaggG, E);
    gemm(HaggIn, attn + N0_W, attn + N0_B, nullptr, T1, Gu, 64, 128, 1);
    gemm(T1, attn + N1_W, attn + N1_B, nullptr, NodeOutG, Gu, 128, 128, 1);
    gemm(ZHin, attn + C0_W, attn + C0_B, nullptr, T1, Gu, 64, 128, 1);
    gemm(T1, attn + C1_W, attn + C1_B, nullptr, ZHG, Gu, 128, 4, 1);

    int e4b = (E * 4 + 255) / 256;
    k_blogit<<<e4b, 256, 0, stream>>>(Rmean, attn, blogit, E);
    k_initmax<<<(NBLK * 4 + 255) / 256, 256, 0, stream>>>(bmax, NBLK * 4);
    hipMemsetAsync(bsum, 0, (size_t)NBLK * 4 * sizeof(float), stream);
    hipMemsetAsync(HaggN, 0, (size_t)N * 128 * sizeof(float), stream);
    hipMemsetAsync(ZaggN, 0, (size_t)N * 12 * sizeof(float), stream);
    k_bmax<<<e4b, 256, 0, stream>>>(blogit, esrc, bmax, E);
    k_bexp<<<e4b, 256, 0, stream>>>(blogit, esrc, bmax, bsum, E);
    k_bnorm<<<e4b, 256, 0, stream>>>(blogit, esrc, bsum, E);
    k_scatterH<<<(int)(((size_t)Gu * 128 + 255) / 256), 256, 0, stream>>>(
        NodeOutG, blogit, esrc, HaggN, Gu);
    k_scatterZ<<<(Gu * 12 + 255) / 256, 256, 0, stream>>>(ZaggG, ZHG, blogit, esrc,
                                                          ZaggN, Gu);
    gemm(HaggN, attn + NO_W, attn + NO_B, curH, nxtH, N, 128, 128, 0);
    k_zupd<<<(N * 3 + 255) / 256, 256, 0, stream>>>(curZ, ZaggN, nxtZ, N);
    { float* t = curH; curH = nxtH; nxtH = t; t = curZ; curZ = nxtZ; nxtZ = t; }

    // ---- ln1 ----
    k_zmean<<<NBATCH, 256, 0, stream>>>(curZ, Zbm, N);
    k_std<<<NBATCH, 256, 0, stream>>>(curZ, Zbm, ln1p, rescB, rrb, N);
    gemm(rrb, ln1p + F0_W, ln1p + F0_B, nullptr, fuse_h, NBATCH, 16, 128, 1);
    gemm(fuse_h, ln1p + F1_W, ln1p + F1_B, nullptr, fuse_o, NBATCH, 128, 128, 1);
    k_lnapply<<<N, 128, 0, stream>>>(curH, curZ, fuse_o, Zbm, rescB, ln1p, nxtH, nxtZ, N);
    { float* t = curH; curH = nxtH; nxtH = t; t = curZ; curZ = nxtZ; nxtZ = t; }

    // ---- ffn ----
    k_blockmean<<<NBLK, 128, 0, stream>>>(curH, curZ, Hblk, Zblk);
    k_ffnin<<<N, 64, 0, stream>>>(curH, curZ, Hblk, Zblk, in272, N);
    gemm(in272, ffnp + H0_W, ffnp + H0_B, nullptr, t256a, N, 272, 256, 1);
    gemm(t256a, ffnp + H1_W, ffnp + H1_B, nullptr, t256b, N, 256, 256, 1);
    gemm(t256b, ffnp + H2_W, ffnp + H2_B, curH, nxtH, N, 256, 128, 0);
    gemm(in272, ffnp + Z0_W, ffnp + Z0_B, nullptr, t256a, N, 272, 256, 1);
    gemm(t256a, ffnp + Z1_W, ffnp + Z1_B, nullptr, t256b, N, 256, 256, 1);
    k_zscal<<<(N + 255) / 256, 256, 0, stream>>>(t256b, ffnp, curZ, Zblk, nxtZ, N);
    { float* t = curH; curH = nxtH; nxtH = t; t = curZ; curZ = nxtZ; nxtZ = t; }

    // ---- ln2 ----
    k_zmean<<<NBATCH, 256, 0, stream>>>(curZ, Zbm, N);
    k_std<<<NBATCH, 256, 0, stream>>>(curZ, Zbm, ln2p, rescB, rrb, N);
    gemm(rrb, ln2p + F0_W, ln2p + F0_B, nullptr, fuse_h, NBATCH, 16, 128, 1);
    gemm(fuse_h, ln2p + F1_W, ln2p + F1_B, nullptr, fuse_o, NBATCH, 128, 128, 1);
    k_lnapply<<<N, 128, 0, stream>>>(curH, curZ, fuse_o, Zbm, rescB, ln2p, nxtH, nxtZ, N);
    { float* t = curH; curH = nxtH; nxtH = t; t = curZ; curZ = nxtZ; nxtZ = t; }
  }

  hipMemcpyAsync(d_out, curH, (size_t)N * 128 * sizeof(float),
                 hipMemcpyDeviceToDevice, stream);
  hipMemcpyAsync((float*)d_out + (size_t)N * 128, curZ, (size_t)N * 3 * sizeof(float),
                 hipMemcpyDeviceToDevice, stream);
}
